// Ranking_8263517078009
// MI455X (gfx1250) — compile-verified, matrix operand order checked by reading, not run on hardware
//
#include <hip/hip_runtime.h>

typedef __attribute__((ext_vector_type(8))) int v8i;

#define NUM_SAMPLES 1000
#define BATCH 16
#define N_ITEMS 2048
#define SIGMA 0.1f
#define CHUNK 8                       // samples per block
#define THREADS 512                   // 16 waves (wave32); 512*4 == N_ITEMS
#define WAVES (THREADS / 32)
#define TPW (N_ITEMS / 16 / WAVES)    // i-tiles per wave = 8

__global__ void zero_kernel(float* p, int n) {
  int i = blockIdx.x * blockDim.x + threadIdx.x;
  if (i < n) p[i] = 0.0f;
}

__global__ __launch_bounds__(THREADS)
void rank_wmma_kernel(const float* __restrict__ x,
                      const float* __restrict__ noise,
                      float* __restrict__ out) {
  __shared__ __align__(16) float srow[N_ITEMS];

  const int b    = blockIdx.y;
  const int s0   = blockIdx.x * CHUNK;
  const int tid  = (int)threadIdx.x;
  const int lane = tid & 31;
  const int wave = tid >> 5;
  const int half = lane >> 4;   // which K-half of the A tile this lane holds
  const int m    = lane & 15;   // A-matrix row (M) within the 16-row tile

  // i32 accumulators: 8 tiles x 8 VGPRs, accumulate across j-chunks AND samples
  v8i acc[TPW];
#pragma unroll
  for (int t = 0; t < TPW; ++t)
#pragma unroll
    for (int e = 0; e < 8; ++e) acc[t][e] = 0;

  // B = all-ones (64x16 IU8): D[m][n] = sum_k A[m][k], layout-independent
  v8i bones;
#pragma unroll
  for (int e = 0; e < 8; ++e) bones[e] = 0x01010101;

  // Each thread owns one float4 of the 2048-wide row. x stays in registers.
  const float* xp = x + (size_t)b * N_ITEMS;
  const float4 xv = *(const float4*)(xp + 4 * tid);
  const float* nbase = noise + (size_t)b * N_ITEMS + 4 * tid;

  // Prime the pipeline: stage sample s0's perturbed row into LDS.
  {
    const float4 nv = *(const float4*)(nbase + (size_t)s0 * BATCH * N_ITEMS);
    float4 pv;
    pv.x = xv.x + SIGMA * nv.x;  pv.y = xv.y + SIGMA * nv.y;
    pv.z = xv.z + SIGMA * nv.z;  pv.w = xv.w + SIGMA * nv.w;
    *(float4*)&srow[4 * tid] = pv;
  }
  __syncthreads();

  for (int ss = 0; ss < CHUNK; ++ss) {
    // Software pipeline: issue next sample's global load before compute;
    // the ~23k VALU instructions below hide the HBM latency.
    const bool have_next = (ss + 1) < CHUNK;
    float4 nxt;
    if (have_next)
      nxt = *(const float4*)(nbase + (size_t)(s0 + ss + 1) * BATCH * N_ITEMS);
    if ((ss + 2) < CHUNK)
      __builtin_prefetch(nbase + (size_t)(s0 + ss + 2) * BATCH * N_ITEMS, 0, 0);

    // per-tile pivot values v[i] for this lane's row M
    float vi[TPW];
#pragma unroll
    for (int t = 0; t < TPW; ++t)
      vi[t] = srow[(wave * TPW + t) * 16 + m];

    for (int jc = 0; jc < N_ITEMS / 64; ++jc) {
      // This lane's 32 j-values for the 16x64 IU8 A layout:
      // K = g*16 + half*8 + (0..7), g = 0..3  -> 8x ds_load_b128 (lane-broadcast)
      const int jb = jc * 64 + half * 8;
      float jv[32];
#pragma unroll
      for (int g = 0; g < 4; ++g) {
        float4 lo = *(const float4*)&srow[jb + g * 16];
        float4 hi = *(const float4*)&srow[jb + g * 16 + 4];
        jv[g*8+0]=lo.x; jv[g*8+1]=lo.y; jv[g*8+2]=lo.z; jv[g*8+3]=lo.w;
        jv[g*8+4]=hi.x; jv[g*8+5]=hi.y; jv[g*8+6]=hi.z; jv[g*8+7]=hi.w;
      }
#pragma unroll
      for (int t = 0; t < TPW; ++t) {
        v8i A;
#pragma unroll
        for (int d = 0; d < 8; ++d) {
          // A VGPR d holds K = (d>>1)*16 + half*8 + (d&1)*4 + (0..3),
          // i.e. jv[(d>>1)*8 + (d&1)*4 + (0..3)], byte0 = lowest K
          const int p = (d >> 1) * 8 + (d & 1) * 4;
          unsigned a0 = (jv[p+0] < vi[t]) ? 1u : 0u;
          unsigned a1 = (jv[p+1] < vi[t]) ? 1u : 0u;
          unsigned a2 = (jv[p+2] < vi[t]) ? 1u : 0u;
          unsigned a3 = (jv[p+3] < vi[t]) ? 1u : 0u;
          A[d] = (int)(a0 | (a1 << 8) | (a2 << 16) | (a3 << 24));
        }
        acc[t] = __builtin_amdgcn_wmma_i32_16x16x64_iu8(
            false, A, false, bones, acc[t], false, false);
      }
    }

    // Publish next sample's row (uniform branch; EXEC stays all-ones for WMMA).
    if (have_next) {
      __syncthreads();
      float4 pv;
      pv.x = xv.x + SIGMA * nxt.x;  pv.y = xv.y + SIGMA * nxt.y;
      pv.z = xv.z + SIGMA * nxt.z;  pv.w = xv.w + SIGMA * nxt.w;
      *(float4*)&srow[4 * tid] = pv;
      __syncthreads();
    }
  }

  // Epilogue: C/D layout — VGPR r, lane L holds D[M = r + 8*(L>=16)][N = L&15],
  // identical across N. Lanes with (lane&15)<8 each write one row.
  const float scale = 1.0f / (float)NUM_SAMPLES;
#pragma unroll
  for (int t = 0; t < TPW; ++t) {
    int cv = 0;
#pragma unroll
    for (int r = 0; r < 8; ++r)
      cv = ((lane & 7) == r) ? acc[t][r] : cv;
    if (m < 8) {
      const int row = (wave * TPW + t) * 16 + (lane & 7) + 8 * half;
      atomicAdd(&out[b * N_ITEMS + row], (float)cv * scale);
    }
  }
}

extern "C" void kernel_launch(void* const* d_in, const int* in_sizes, int n_in,
                              void* d_out, int out_size, void* d_ws, size_t ws_size,
                              hipStream_t stream) {
  const float* x     = (const float*)d_in[0];   // [16, 2048] f32
  const float* noise = (const float*)d_in[1];   // [1000, 16, 2048] f32
  float* out = (float*)d_out;                   // [16, 2048] f32

  // d_out is poisoned by the harness; zero it before atomic accumulation.
  zero_kernel<<<(out_size + 255) / 256, 256, 0, stream>>>(out, out_size);

  dim3 grid(NUM_SAMPLES / CHUNK, BATCH);        // 125 x 16 = 2000 blocks
  rank_wmma_kernel<<<grid, THREADS, 0, stream>>>(x, noise, out);
}
